// TimeOnlyMambaBlock_32315333935258
// MI455X (gfx1250) — matrix operand order repaired and checked by reading, not compiled, not run on hardware
//
#include <hip/hip_runtime.h>
#include <hip/hip_bf16.h>
#include <math.h>

// ---------------------------------------------------------------------------
// TimeOnlyMambaBlock fused pipeline for MI455X (gfx1250, wave32, WMMA)
// Round 3: 32x64 wave tiles (8 WMMA/k-step) + chunk-parallel selective scan.
// ---------------------------------------------------------------------------
#define B_SZ    4
#define L_SEQ   2048
#define D_MODEL 256
#define D_INNER 512
#define D_STATE 16
#define DT_RANK 16
#define XPROJ_N 48
#define M_ROWS  (B_SZ * L_SEQ)   // 8192
#define EPSV    1e-5f
#define KPAD    264               // 256 f16 + 8 f16 pad (16B) -> bank spread
#define TCHUNK  64                // scan chunk length
#define NCHUNK  (L_SEQ / TCHUNK)  // 32
#define NCH     (B_SZ * D_INNER)  // 2048 channels

typedef __attribute__((ext_vector_type(16))) _Float16 v16h;
typedef __attribute__((ext_vector_type(8)))  _Float16 v8h;
typedef __attribute__((ext_vector_type(4)))  _Float16 v4h;
typedef __attribute__((ext_vector_type(8)))  float    v8f;

// A fragment (16x32 f16): 16 consecutive K values at p
__device__ __forceinline__ v16h ldfragA(const _Float16* __restrict__ p) {
    v8h lo = *(const v8h*)p;
    v8h hi = *(const v8h*)(p + 8);
    return __builtin_shufflevector(lo, hi, 0,1,2,3,4,5,6,7,8,9,10,11,12,13,14,15);
}
// B fragment (32x16 f16): 16 consecutive K values at p
__device__ __forceinline__ v16h ldfragB(const _Float16* __restrict__ p) {
    v8h lo = *(const v8h*)p;
    v8h hi = *(const v8h*)(p + 8);
    return __builtin_shufflevector(lo, hi, 0,1,2,3,4,5,6,7,8,9,10,11,12,13,14,15);
}

__device__ __forceinline__ void store_c_frag(float* __restrict__ out, int ldn,
                                             int rowBase, int colBase,
                                             int lane, const v8f& acc) {
    const int half = lane >> 4;
    const int col  = colBase + (lane & 15);
#pragma unroll
    for (int r = 0; r < 8; ++r)
        out[(size_t)(rowBase + r + half * 8) * ldn + col] = acc[r];
}

#define WMMA16(a, b, c) __builtin_amdgcn_wmma_f32_16x16x32_f16(false, a, false, b, (short)0, c, false, false)

// ---------------------------------------------------------------------------
// fp32 -> f16 bulk convert (4 elems/thread, n % 4 == 0)
// ---------------------------------------------------------------------------
__global__ void k_cvt(const float* __restrict__ s, _Float16* __restrict__ d, int n) {
    const int i = (blockIdx.x * 256 + threadIdx.x) * 4;
    if (i < n) {
        const float4 v = *(const float4*)(s + i);
        v4h r;
        r[0] = (_Float16)v.x; r[1] = (_Float16)v.y;
        r[2] = (_Float16)v.z; r[3] = (_Float16)v.w;
        *(v4h*)(d + i) = r;
    }
}

// ---------------------------------------------------------------------------
// Kernel 1: xz = x @ in_proj_w^T  [8192,256]x[1024,256]^T -> [8192,1024] f32
// 8 waves; wave tile 32Mx64N (8 acc); block tile 256Mx64N; B tile in LDS
// ---------------------------------------------------------------------------
__global__ void k_gemm_inproj(const _Float16* __restrict__ X16,
                              const _Float16* __restrict__ W16,
                              float* __restrict__ XZ) {
    __shared__ _Float16 sB[64 * KPAD];
    const int lane = threadIdx.x & 31;
    const int wid  = threadIdx.x >> 5;
    const int mBase = blockIdx.x * 256 + wid * 32;
    const int nBase = blockIdx.y * 64;

    for (int i = threadIdx.x; i < 64 * 32; i += 256) {
        const int n  = i >> 5;
        const int k8 = (i & 31) << 3;
        *(v8h*)&sB[n * KPAD + k8] = *(const v8h*)&W16[(size_t)(nBase + n) * D_MODEL + k8];
    }
    __syncthreads();

    const int laneM = lane & 15;
    const int half  = lane >> 4;
    const _Float16* rowptr0 = X16 + (size_t)(mBase + laneM) * D_MODEL;
    const _Float16* rowptr1 = rowptr0 + (size_t)16 * D_MODEL;

    v8f acc[2][4] = {};
    for (int k0 = 0; k0 < D_MODEL; k0 += 32) {
        const int kb = k0 + (half << 4);
        const int kk = k0 + (half << 4);
        const _Float16* bb = &sB[laneM * KPAD + kk];
        v16h a0 = ldfragA(rowptr0 + kb);
        v16h a1 = ldfragA(rowptr1 + kb);
        v16h b0 = ldfragB(bb);
        v16h b1 = ldfragB(bb + 16 * KPAD);
        v16h b2 = ldfragB(bb + 32 * KPAD);
        v16h b3 = ldfragB(bb + 48 * KPAD);
        acc[0][0] = WMMA16(a0, b0, acc[0][0]);
        acc[1][0] = WMMA16(a1, b0, acc[1][0]);
        acc[0][1] = WMMA16(a0, b1, acc[0][1]);
        acc[1][1] = WMMA16(a1, b1, acc[1][1]);
        acc[0][2] = WMMA16(a0, b2, acc[0][2]);
        acc[1][2] = WMMA16(a1, b2, acc[1][2]);
        acc[0][3] = WMMA16(a0, b3, acc[0][3]);
        acc[1][3] = WMMA16(a1, b3, acc[1][3]);
    }
#pragma unroll
    for (int m2 = 0; m2 < 2; ++m2)
#pragma unroll
        for (int nt = 0; nt < 4; ++nt)
            store_c_frag(XZ, 2 * D_INNER, mBase + m2 * 16, nBase + nt * 16, lane, acc[m2][nt]);
}

// ---------------------------------------------------------------------------
// Kernel 2: causal depthwise conv (k=4) + bias + SiLU; writes fp32 + f16
// ---------------------------------------------------------------------------
__global__ void k_conv_silu(const float* __restrict__ XZ,
                            const float* __restrict__ CW,
                            const float* __restrict__ CB,
                            float* __restrict__ XIN,
                            _Float16* __restrict__ XIN16) {
    const int idx = blockIdx.x * blockDim.x + threadIdx.x;
    if (idx >= M_ROWS * D_INNER) return;
    const int d   = idx & (D_INNER - 1);
    const int row = idx >> 9;
    const int l   = row & (L_SEQ - 1);
    float acc = CB[d];
#pragma unroll
    for (int j = 0; j < 4; ++j) {
        const int ll = l - 3 + j;
        const float v = (ll >= 0) ? XZ[(size_t)(row - 3 + j) * (2 * D_INNER) + d] : 0.0f;
        acc = fmaf(CW[d * 4 + j], v, acc);
    }
    const float s = acc * (1.0f / (1.0f + __expf(-acc)));
    XIN[idx]   = s;
    XIN16[idx] = (_Float16)s;
}

// ---------------------------------------------------------------------------
// Kernel 3: x_dbl = xin @ x_proj_w^T -> [8192,48] f32
// wave tile 32Mx48N; B tile (48 x 256-K chunk) in LDS; two K chunks
// ---------------------------------------------------------------------------
__global__ void k_gemm_xproj(const _Float16* __restrict__ XIN16,
                             const _Float16* __restrict__ W16,
                             float* __restrict__ XD) {
    __shared__ _Float16 sB[48 * KPAD];
    const int lane = threadIdx.x & 31;
    const int wid  = threadIdx.x >> 5;
    const int mBase = blockIdx.x * 256 + wid * 32;
    const int laneM = lane & 15;
    const int half  = lane >> 4;
    const _Float16* rowptr0 = XIN16 + (size_t)(mBase + laneM) * D_INNER;
    const _Float16* rowptr1 = rowptr0 + (size_t)16 * D_INNER;

    v8f acc[2][3] = {};
    for (int kc = 0; kc < D_INNER; kc += 256) {
        __syncthreads();
        for (int i = threadIdx.x; i < 48 * 32; i += 256) {
            const int n  = i >> 5;
            const int k8 = (i & 31) << 3;
            *(v8h*)&sB[n * KPAD + k8] = *(const v8h*)&W16[(size_t)n * D_INNER + kc + k8];
        }
        __syncthreads();
        for (int k0 = 0; k0 < 256; k0 += 32) {
            const int kb = kc + k0 + (half << 4);
            const int kk = k0 + (half << 4);
            const _Float16* bb = &sB[laneM * KPAD + kk];
            v16h a0 = ldfragA(rowptr0 + kb);
            v16h a1 = ldfragA(rowptr1 + kb);
            v16h b0 = ldfragB(bb);
            v16h b1 = ldfragB(bb + 16 * KPAD);
            v16h b2 = ldfragB(bb + 32 * KPAD);
            acc[0][0] = WMMA16(a0, b0, acc[0][0]);
            acc[1][0] = WMMA16(a1, b0, acc[1][0]);
            acc[0][1] = WMMA16(a0, b1, acc[0][1]);
            acc[1][1] = WMMA16(a1, b1, acc[1][1]);
            acc[0][2] = WMMA16(a0, b2, acc[0][2]);
            acc[1][2] = WMMA16(a1, b2, acc[1][2]);
        }
    }
#pragma unroll
    for (int m2 = 0; m2 < 2; ++m2)
#pragma unroll
        for (int nt = 0; nt < 3; ++nt)
            store_c_frag(XD, XPROJ_N, mBase + m2 * 16, nt * 16, lane, acc[m2][nt]);
}

// ---------------------------------------------------------------------------
// Kernel 4: dt = softplus(x_dbl[:, :16] @ dt_proj_w^T + dt_proj_b)   (fp32)
// ---------------------------------------------------------------------------
__global__ void k_gemm_dtproj(const float* __restrict__ XD,
                              const float* __restrict__ W,
                              const float* __restrict__ Bias,
                              float* __restrict__ DT) {
    const int lane = threadIdx.x & 31;
    const int wid  = threadIdx.x >> 5;
    const int mBase = blockIdx.x * 128 + wid * 16;
    const int nBase = blockIdx.y * 64;
    const float* rowptr = XD + (size_t)(mBase + (lane & 15)) * XPROJ_N;

    v16h a;
    {
        const bool lo = (lane < 16);
#pragma unroll
        for (int i = 0; i < 16; ++i) a[i] = lo ? (_Float16)rowptr[i] : (_Float16)0.0f;
    }

    v8f acc[4] = {};
#pragma unroll
    for (int nt = 0; nt < 4; ++nt) {
        v16h b = {};
        if (lane < 16) {
            const float* colptr = W + (size_t)(nBase + nt * 16 + lane) * DT_RANK;
#pragma unroll
            for (int i = 0; i < 16; ++i) b[i] = (_Float16)colptr[i];
        }
        acc[nt] = WMMA16(a, b, acc[nt]);
    }

    const int half = lane >> 4;
#pragma unroll
    for (int nt = 0; nt < 4; ++nt) {
        const int col = nBase + nt * 16 + (lane & 15);
        const float bia = Bias[col];
#pragma unroll
        for (int r = 0; r < 8; ++r) {
            float v = acc[nt][r] + bia;
            v = (v > 20.0f) ? v : log1pf(__expf(v));
            DT[(size_t)(mBase + r + half * 8) * D_INNER + col] = v;
        }
    }
}

// ---------------------------------------------------------------------------
// Chunk-parallel selective scan (3 phases).
// Phase A: per (chunk, channel): local scan from h=0 over 64 steps.
//   Writes y_local (f32), chunk decay exp(sum w) and local end state.
// ---------------------------------------------------------------------------
__global__ void k_scan_local(const float* __restrict__ XIN,
                             const float* __restrict__ DT,
                             const float* __restrict__ XD,
                             const float* __restrict__ A_log,
                             float* __restrict__ YLOC,
                             float* __restrict__ HEND,
                             float* __restrict__ PDEC) {
    __shared__ float sBC[TCHUNK * 32];
    const int chunk = blockIdx.x;
    const int ch    = blockIdx.y * 256 + threadIdx.x;   // 0..2047
    const int b     = ch >> 9;
    const int d     = ch & (D_INNER - 1);
    const int rowBase = b * L_SEQ + chunk * TCHUNK;

    float Aneg[D_STATE], h[D_STATE], wsum[D_STATE];
#pragma unroll
    for (int s = 0; s < D_STATE; ++s) {
        Aneg[s] = -__expf(A_log[d * D_STATE + s]);
        h[s] = 0.0f; wsum[s] = 0.0f;
    }

    for (int i = threadIdx.x; i < TCHUNK * 32; i += 256) {
        const int tt = i >> 5, c = i & 31;
        sBC[i] = XD[(size_t)(rowBase + tt) * XPROJ_N + DT_RANK + c];
    }
    __syncthreads();

    for (int t = 0; t < TCHUNK; ++t) {
        const int row = rowBase + t;
        const float dtv = DT[(size_t)row * D_INNER + d];
        const float xv  = XIN[(size_t)row * D_INNER + d];
        const float* Bm = &sBC[t * 32];
        const float* Cm = Bm + D_STATE;
        const float dbx = dtv * xv;
        float y = 0.0f;
#pragma unroll
        for (int s = 0; s < D_STATE; ++s) {
            const float w = dtv * Aneg[s];
            wsum[s] += w;
            h[s] = fmaf(__expf(w), h[s], dbx * Bm[s]);
            y = fmaf(h[s], Cm[s], y);
        }
        YLOC[(size_t)row * D_INNER + d] = y;
    }

    const size_t base = ((size_t)chunk * NCH + ch) * D_STATE;
#pragma unroll
    for (int s = 0; s < D_STATE; ++s) {
        HEND[base + s] = h[s];
        PDEC[base + s] = __expf(wsum[s]);
    }
}

// ---------------------------------------------------------------------------
// Phase B: sequential combine of chunk carries. Thread = (channel, state).
// ---------------------------------------------------------------------------
__global__ void k_scan_carry(const float* __restrict__ HEND,
                             const float* __restrict__ PDEC,
                             float* __restrict__ HSTART) {
    const int idx = blockIdx.x * 256 + threadIdx.x;     // 0..32767
    if (idx >= NCH * D_STATE) return;
    float h = 0.0f;
    for (int c = 0; c < NCHUNK; ++c) {
        const size_t o = (size_t)c * NCH * D_STATE + idx;
        HSTART[o] = h;
        h = fmaf(PDEC[o], h, HEND[o]);
    }
}

// ---------------------------------------------------------------------------
// Phase C: fixup + D-skip + SiLU gate -> f16 output for out_proj.
//   y_t = y_local_t + C_t . (exp(cumw_t) * h_start)
// ---------------------------------------------------------------------------
__global__ void k_scan_fix(const float* __restrict__ XIN,
                           const float* __restrict__ XZ,
                           const float* __restrict__ DT,
                           const float* __restrict__ XD,
                           const float* __restrict__ A_log,
                           const float* __restrict__ Dp,
                           const float* __restrict__ YLOC,
                           const float* __restrict__ HSTART,
                           _Float16* __restrict__ YG16) {
    __shared__ float sBC[TCHUNK * 32];
    const int chunk = blockIdx.x;
    const int ch    = blockIdx.y * 256 + threadIdx.x;
    const int b     = ch >> 9;
    const int d     = ch & (D_INNER - 1);
    const int rowBase = b * L_SEQ + chunk * TCHUNK;

    float Aneg[D_STATE], hs[D_STATE], cumw[D_STATE];
    const size_t base = ((size_t)chunk * NCH + ch) * D_STATE;
#pragma unroll
    for (int s = 0; s < D_STATE; ++s) {
        Aneg[s] = -__expf(A_log[d * D_STATE + s]);
        hs[s]   = HSTART[base + s];
        cumw[s] = 0.0f;
    }
    const float dp = Dp[d];

    for (int i = threadIdx.x; i < TCHUNK * 32; i += 256) {
        const int tt = i >> 5, c = i & 31;
        sBC[i] = XD[(size_t)(rowBase + tt) * XPROJ_N + DT_RANK + c];
    }
    __syncthreads();

    for (int t = 0; t < TCHUNK; ++t) {
        const int row = rowBase + t;
        const float dtv = DT[(size_t)row * D_INNER + d];
        const float xv  = XIN[(size_t)row * D_INNER + d];
        const float zv  = XZ[(size_t)row * (2 * D_INNER) + D_INNER + d];
        const float* Cm = &sBC[t * 32 + D_STATE];
        float y = YLOC[(size_t)row * D_INNER + d];
#pragma unroll
        for (int s = 0; s < D_STATE; ++s) {
            cumw[s] += dtv * Aneg[s];
            y = fmaf(__expf(cumw[s]) * hs[s], Cm[s], y);
        }
        y = fmaf(xv, dp, y);
        const float sig = 1.0f / (1.0f + __expf(-zv));
        YG16[(size_t)row * D_INNER + d] = (_Float16)(y * (zv * sig));
    }
}

// ---------------------------------------------------------------------------
// Kernel 6: y_out = yg @ out_proj_w^T -> [8192,256] f32
// wave tile 32Mx64N; B tile in LDS (two 256-K chunks)
// ---------------------------------------------------------------------------
__global__ void k_gemm_outproj(const _Float16* __restrict__ YG16,
                               const _Float16* __restrict__ W16,
                               float* __restrict__ YO) {
    __shared__ _Float16 sB[64 * KPAD];
    const int lane = threadIdx.x & 31;
    const int wid  = threadIdx.x >> 5;
    const int mBase = blockIdx.x * 256 + wid * 32;
    const int nBase = blockIdx.y * 64;
    const int laneM = lane & 15;
    const int half  = lane >> 4;
    const _Float16* rowptr0 = YG16 + (size_t)(mBase + laneM) * D_INNER;
    const _Float16* rowptr1 = rowptr0 + (size_t)16 * D_INNER;

    v8f acc[2][4] = {};
    for (int kc = 0; kc < D_INNER; kc += 256) {
        __syncthreads();
        for (int i = threadIdx.x; i < 64 * 32; i += 256) {
            const int n  = i >> 5;
            const int k8 = (i & 31) << 3;
            *(v8h*)&sB[n * KPAD + k8] = *(const v8h*)&W16[(size_t)(nBase + n) * D_INNER + kc + k8];
        }
        __syncthreads();
        for (int k0 = 0; k0 < 256; k0 += 32) {
            const int kb = kc + k0 + (half << 4);
            const int kk = k0 + (half << 4);
            const _Float16* bb = &sB[laneM * KPAD + kk];
            v16h a0 = ldfragA(rowptr0 + kb);
            v16h a1 = ldfragA(rowptr1 + kb);
            v16h b0 = ldfragB(bb);
            v16h b1 = ldfragB(bb + 16 * KPAD);
            v16h b2 = ldfragB(bb + 32 * KPAD);
            v16h b3 = ldfragB(bb + 48 * KPAD);
            acc[0][0] = WMMA16(a0, b0, acc[0][0]);
            acc[1][0] = WMMA16(a1, b0, acc[1][0]);
            acc[0][1] = WMMA16(a0, b1, acc[0][1]);
            acc[1][1] = WMMA16(a1, b1, acc[1][1]);
            acc[0][2] = WMMA16(a0, b2, acc[0][2]);
            acc[1][2] = WMMA16(a1, b2, acc[1][2]);
            acc[0][3] = WMMA16(a0, b3, acc[0][3]);
            acc[1][3] = WMMA16(a1, b3, acc[1][3]);
        }
    }
#pragma unroll
    for (int m2 = 0; m2 < 2; ++m2)
#pragma unroll
        for (int nt = 0; nt < 4; ++nt)
            store_c_frag(YO, D_MODEL, mBase + m2 * 16, nBase + nt * 16, lane, acc[m2][nt]);
}

// ---------------------------------------------------------------------------
// Kernel 7: x1 = LN(y); x2 = LN(x + x1). One wave per row.
// ---------------------------------------------------------------------------
__device__ __forceinline__ float wave_sum(float v) {
#pragma unroll
    for (int off = 16; off >= 1; off >>= 1) v += __shfl_xor(v, off);
    return v;
}

__global__ void k_ln(const float* __restrict__ Y,
                     const float* __restrict__ X,
                     const float* __restrict__ g1, const float* __restrict__ b1,
                     const float* __restrict__ g2, const float* __restrict__ b2,
                     float* __restrict__ X2) {
    const int wid  = threadIdx.x >> 5;
    const int lane = threadIdx.x & 31;
    const int row  = blockIdx.x * 8 + wid;
    const float* yr = Y + (size_t)row * D_MODEL;
    const float* xr = X + (size_t)row * D_MODEL;

    float v[8];
    float s = 0.0f, ss = 0.0f;
#pragma unroll
    for (int i = 0; i < 8; ++i) {
        const float t = yr[lane + i * 32];
        v[i] = t; s += t; ss = fmaf(t, t, ss);
    }
    s = wave_sum(s); ss = wave_sum(ss);
    const float mean = s * (1.0f / D_MODEL);
    const float inv  = rsqrtf(ss * (1.0f / D_MODEL) - mean * mean + EPSV);

    float w[8];
    float s2 = 0.0f, ss2 = 0.0f;
#pragma unroll
    for (int i = 0; i < 8; ++i) {
        const int c = lane + i * 32;
        const float x1 = (v[i] - mean) * inv * g1[c] + b1[c];
        const float t  = xr[c] + x1;
        w[i] = t; s2 += t; ss2 = fmaf(t, t, ss2);
    }
    s2 = wave_sum(s2); ss2 = wave_sum(ss2);
    const float mean2 = s2 * (1.0f / D_MODEL);
    const float inv2  = rsqrtf(ss2 * (1.0f / D_MODEL) - mean2 * mean2 + EPSV);
#pragma unroll
    for (int i = 0; i < 8; ++i) {
        const int c = lane + i * 32;
        X2[(size_t)row * D_MODEL + c] = (w[i] - mean2) * inv2 * g2[c] + b2[c];
    }
}

// ---------------------------------------------------------------------------
// Kernel 8: grouped conv (groups=4,k=3,pad=1) + BN(eval) + GELU + residual
// ---------------------------------------------------------------------------
__global__ void k_enh(const float* __restrict__ X2,
                      const float* __restrict__ EW,
                      const float* __restrict__ EB,
                      const float* __restrict__ BNg, const float* __restrict__ BNb,
                      const float* __restrict__ BNm, const float* __restrict__ BNv,
                      float* __restrict__ OUT) {
    __shared__ float tile[18 * D_MODEL];
    const int b  = blockIdx.y;
    const int l0 = blockIdx.x * 16;
    const int c  = threadIdx.x;

    for (int i = threadIdx.x; i < 18 * D_MODEL; i += 256) {
        const int rr = i >> 8;
        const int cc = i & (D_MODEL - 1);
        const int l  = l0 - 1 + rr;
        tile[i] = (l >= 0 && l < L_SEQ) ? X2[(size_t)(b * L_SEQ + l) * D_MODEL + cc] : 0.0f;
    }
    __syncthreads();

    float acc[16];
    const float bia = EB[c];
#pragma unroll
    for (int li = 0; li < 16; ++li) acc[li] = bia;

    const int icb = (c >> 6) << 6;
    for (int j = 0; j < 3; ++j) {
        for (int ic = 0; ic < 64; ++ic) {
            const float wv = EW[c * 192 + ic * 3 + j];
            const float* src = &tile[j * D_MODEL + icb + ic];
#pragma unroll
            for (int li = 0; li < 16; ++li)
                acc[li] = fmaf(wv, src[li * D_MODEL], acc[li]);
        }
    }

    const float mean = BNm[c];
    const float inv  = rsqrtf(BNv[c] + EPSV);
    const float gg   = BNg[c], bb = BNb[c];
#pragma unroll
    for (int li = 0; li < 16; ++li) {
        float v = (acc[li] - mean) * inv * gg + bb;
        const float gel = 0.5f * v * (1.0f + erff(v * 0.70710678118f));
        OUT[(size_t)(b * L_SEQ + l0 + li) * D_MODEL + c] = tile[(li + 1) * D_MODEL + c] + gel;
    }
}

// ---------------------------------------------------------------------------
extern "C" void kernel_launch(void* const* d_in, const int* in_sizes, int n_in,
                              void* d_out, int out_size, void* d_ws, size_t ws_size,
                              hipStream_t stream) {
    (void)in_sizes; (void)n_in; (void)out_size; (void)ws_size;
    const float* x         = (const float*)d_in[0];
    const float* in_proj_w = (const float*)d_in[1];
    const float* conv_w    = (const float*)d_in[2];
    const float* conv_b    = (const float*)d_in[3];
    const float* x_proj_w  = (const float*)d_in[4];
    const float* dt_proj_w = (const float*)d_in[5];
    const float* dt_proj_b = (const float*)d_in[6];
    const float* A_log     = (const float*)d_in[7];
    const float* Dp        = (const float*)d_in[8];
    const float* out_proj_w= (const float*)d_in[9];
    const float* ln1_g     = (const float*)d_in[10];
    const float* ln1_b     = (const float*)d_in[11];
    const float* ln2_g     = (const float*)d_in[12];
    const float* ln2_b     = (const float*)d_in[13];
    const float* enh_w     = (const float*)d_in[14];
    const float* enh_b     = (const float*)d_in[15];
    const float* bn_g      = (const float*)d_in[16];
    const float* bn_b      = (const float*)d_in[17];
    const float* bn_mean   = (const float*)d_in[18];
    const float* bn_var    = (const float*)d_in[19];
    float* out = (float*)d_out;

    // fp32 workspace
    float* xz     = (float*)d_ws;                         // [8192,1024]
    float* xin    = xz     + (size_t)M_ROWS * 2 * D_INNER;// [8192,512]
    float* xdbl   = xin    + (size_t)M_ROWS * D_INNER;    // [8192,48]
    float* dt     = xdbl   + (size_t)M_ROWS * XPROJ_N;    // [8192,512]
    float* yo     = dt     + (size_t)M_ROWS * D_INNER;    // [8192,256]
    float* x2     = yo     + (size_t)M_ROWS * D_MODEL;    // [8192,256]
    float* yloc   = x2     + (size_t)M_ROWS * D_MODEL;    // [8192,512]
    float* hend   = yloc   + (size_t)M_ROWS * D_INNER;    // [32,2048,16]
    float* pdec   = hend   + (size_t)NCHUNK * NCH * D_STATE;
    float* hstart = pdec   + (size_t)NCHUNK * NCH * D_STATE;
    // f16 workspace
    _Float16* x16   = (_Float16*)(hstart + (size_t)NCHUNK * NCH * D_STATE);
    _Float16* xin16 = x16   + (size_t)M_ROWS * D_MODEL;       // [8192,512]
    _Float16* yg16  = xin16 + (size_t)M_ROWS * D_INNER;       // [8192,512]
    _Float16* wi16  = yg16  + (size_t)M_ROWS * D_INNER;       // [1024,256]
    _Float16* wx16  = wi16  + (size_t)(2 * D_INNER) * D_MODEL;// [48,512]
    _Float16* wo16  = wx16  + (size_t)XPROJ_N * D_INNER;      // [256,512]

    // 0) one-time fp32 -> f16 conversions
    k_cvt<<<(M_ROWS * D_MODEL) / 1024, 256, 0, stream>>>(x, x16, M_ROWS * D_MODEL);
    k_cvt<<<(2 * D_INNER * D_MODEL) / 1024, 256, 0, stream>>>(in_proj_w, wi16, 2 * D_INNER * D_MODEL);
    k_cvt<<<(XPROJ_N * D_INNER) / 1024, 256, 0, stream>>>(x_proj_w, wx16, XPROJ_N * D_INNER);
    k_cvt<<<(D_MODEL * D_INNER) / 1024, 256, 0, stream>>>(out_proj_w, wo16, D_MODEL * D_INNER);

    k_gemm_inproj <<<dim3(M_ROWS / 256, (2 * D_INNER) / 64), 256, 0, stream>>>(x16, wi16, xz);
    k_conv_silu   <<<(M_ROWS * D_INNER) / 256, 256, 0, stream>>>(xz, conv_w, conv_b, xin, xin16);
    k_gemm_xproj  <<<dim3(M_ROWS / 256, 1), 256, 0, stream>>>(xin16, wx16, xdbl);
    k_gemm_dtproj <<<dim3(M_ROWS / 128, D_INNER / 64), 256, 0, stream>>>(xdbl, dt_proj_w, dt_proj_b, dt);

    // chunk-parallel selective scan
    k_scan_local<<<dim3(NCHUNK, NCH / 256), 256, 0, stream>>>(xin, dt, xdbl, A_log, yloc, hend, pdec);
    k_scan_carry<<<(NCH * D_STATE) / 256, 256, 0, stream>>>(hend, pdec, hstart);
    k_scan_fix  <<<dim3(NCHUNK, NCH / 256), 256, 0, stream>>>(xin, xz, dt, xdbl, A_log, Dp, yloc, hstart, yg16);

    k_gemm_outproj<<<dim3(M_ROWS / 256, D_MODEL / 64), 256, 0, stream>>>(yg16, wo16, yo);
    k_ln          <<<M_ROWS / 8, 256, 0, stream>>>(yo, x, ln1_g, ln1_b, ln2_g, ln2_b, x2);
    k_enh         <<<dim3(L_SEQ / 16, B_SZ), 256, 0, stream>>>(x2, enh_w, enh_b, bn_g, bn_b, bn_mean, bn_var, out);
}
